// LSTMTransparent_54666343743621
// MI455X (gfx1250) — compile-verified, matrix operand order checked by reading, not run on hardware
//
#include <hip/hip_runtime.h>

// ---------------------------------------------------------------------------
// LSTM forward for MI455X (gfx1250, wave32) — persistent-kernel version.
//
// Serial recurrence (T=512 steps of a 64x4096x1024 GEMM + cell update) is
// latency-bound: weights (8.4MB bf16 each) are L2-resident (192MB L2), per-step
// math is ~1us, so per-step kernel-launch latency would dominate. Instead:
// ONE persistent kernel, 32 workgroups x 256 threads, loops over t internally
// with a device-scope phase barrier between steps.
//  - bf16 WMMA (v_wmma_f32_16x16x32_bf16), fp32 accumulate.
//  - Each wave owns the i,f,g,o gate tiles for the SAME (m,n) -> lane-local
//    cell update; cell state c lives in registers for all 512 steps.
//  - h double-buffered in ws (bf16); x pre-converted to bf16 by prep kernels.
//  - Block 0 prefetches x(t+1) (global_prefetch_b8); others hit L2.
// ---------------------------------------------------------------------------

typedef __attribute__((ext_vector_type(16))) __bf16 v16bf;
typedef __attribute__((ext_vector_type(8)))  __bf16 v8bf;
typedef __attribute__((ext_vector_type(8)))  float  v8f;

constexpr int B = 64;
constexpr int T = 512;
constexpr int I = 1024;
constexpr int H = 1024;
constexpr int G = 4 * H;      // 4096 gate columns
constexpr int NBLK = 32;      // persistent grid size

// One 16x32 bf16 WMMA operand fragment from a per-lane pointer (two 16B chunks
// at +0 and +32 bytes, per the documented 16-bit A/B VGPR layout).
__device__ __forceinline__ v16bf ldfrag(const __bf16* __restrict__ p) {
  union { v16bf v; v8bf h[2]; } u;
  u.h[0] = *reinterpret_cast<const v8bf*>(p);
  u.h[1] = *reinterpret_cast<const v8bf*>(p + 16);
  return u.v;
}

__device__ __forceinline__ v8f wmma_bf16(v16bf a, v16bf b, v8f c) {
  return __builtin_amdgcn_wmma_f32_16x16x32_bf16(false, a, false, b,
                                                 (short)0, c, false, false);
}

__device__ __forceinline__ float fast_tanh(float x) {
#if __has_builtin(__builtin_amdgcn_tanhf)
  return __builtin_amdgcn_tanhf(x);
#elif __has_builtin(__builtin_amdgcn_tanh_f32)
  return __builtin_amdgcn_tanh_f32(x);
#else
  return tanhf(x);
#endif
}
__device__ __forceinline__ float fast_sigmoid(float x) {
  return 0.5f + 0.5f * fast_tanh(0.5f * x);   // one transcendental
}

// Device-wide sense/phase barrier. bar[0]=arrival counter, bar[1]=phase.
__device__ __forceinline__ void grid_sync(unsigned* bar, unsigned target) {
  __threadfence();                 // release this block's writes (agent scope)
  __syncthreads();
  if (threadIdx.x == 0) {
    unsigned* cnt   = bar;
    unsigned* phase = bar + 1;
    const unsigned prev = __hip_atomic_fetch_add(cnt, 1u, __ATOMIC_ACQ_REL,
                                                 __HIP_MEMORY_SCOPE_AGENT);
    if (prev == (unsigned)(NBLK - 1)) {
      __hip_atomic_store(cnt, 0u, __ATOMIC_RELAXED, __HIP_MEMORY_SCOPE_AGENT);
      __hip_atomic_fetch_add(phase, 1u, __ATOMIC_RELEASE,
                             __HIP_MEMORY_SCOPE_AGENT);
    } else {
      while (__hip_atomic_load(phase, __ATOMIC_ACQUIRE,
                               __HIP_MEMORY_SCOPE_AGENT) < target)
        __builtin_amdgcn_s_sleep(8);
    }
  }
  __syncthreads();
  __threadfence();                 // acquire side for all threads in block
}

// ---------------------------------------------------------------------------
// Persistent LSTM kernel. Grid: 32 blocks x 256 threads (8 wave32s).
//  block b owns hidden columns [b*32, b*32+32)
//  wave w: m-tile = (w>>1)*16 (batch rows), n-subtile = (w&1)*16
// ---------------------------------------------------------------------------
__global__ __launch_bounds__(256) void lstm_persistent_kernel(
    const __bf16* __restrict__ xb,    // [B][T][I] bf16
    const __bf16* __restrict__ wih,   // [G][I]    bf16
    const __bf16* __restrict__ whh,   // [G][H]    bf16
    const float*  __restrict__ bsum,  // [G]       b_ih + b_hh
    __bf16*       __restrict__ h0,    // [B][H]    bf16 state, parity 0
    __bf16*       __restrict__ h1,    // [B][H]    bf16 state, parity 1
    float*        __restrict__ out,   // [B][T][H]
    float*        __restrict__ hn,    // [B][H]
    float*        __restrict__ cn,    // [B][H]
    unsigned*     __restrict__ bar)   // barrier state (cnt, phase)
{
  const int lane = threadIdx.x & 31;
  const int wave = threadIdx.x >> 5;
  const int m0 = (wave >> 1) * 16;                  // batch-row tile
  const int n0 = blockIdx.x * 32 + (wave & 1) * 16; // hidden-col tile
  const int r  = lane & 15;                         // fragment row within tile
  const int ko = (lane >> 4) << 3;                  // 0 or 8 (K sub-offset)

  // Hoisted per-lane fragment base pointers (loop-invariant except x row += I).
  const __bf16* pA1 = xb + (size_t)(m0 + r) * (T * I) + ko;
  const __bf16* pB1[4];
  const __bf16* pB2[4];
#pragma unroll
  for (int g = 0; g < 4; ++g) {
    pB1[g] = wih + (size_t)(g * H + n0 + r) * I + ko;
    pB2[g] = whh + (size_t)(g * H + n0 + r) * H + ko;
  }
  const __bf16* pH0 = h0 + (size_t)(m0 + r) * H + ko;
  const __bf16* pH1 = h1 + (size_t)(m0 + r) * H + ko;

  // Lane-local output element mapping (C/D layout):
  //   VGPR e -> M = m0 + (lane<16 ? e : e+8), N = n0 + (lane&15)
  const int n     = n0 + r;
  const int mbase = m0 + ((lane >> 4) << 3);
  const float bi = bsum[n];
  const float bf = bsum[H + n];
  const float bg = bsum[2 * H + n];
  const float bo = bsum[3 * H + n];

  float creg[8];                    // cell state lives in registers all T steps
#pragma unroll
  for (int e = 0; e < 8; ++e) creg[e] = 0.0f;

  for (int t = 0; t < T; ++t) {
    const __bf16* pa2 = (t & 1) ? pH1 : pH0;   // h_{t-1} fragment base
    __bf16*       hx  = (t & 1) ? h0 : h1;     // h_t buffer

    // Block 0 prefetches next step's x tile into L2 (others reuse from L2).
    if (blockIdx.x == 0 && (t + 1) < T) {
      const char* xn = (const char*)xb + (size_t)(t + 1) * I * sizeof(__bf16);
#pragma unroll
      for (int j = 0; j < 4; ++j) {
        const int l   = ((int)threadIdx.x << 2) + j;   // 1024 cache lines
        const int mm  = l >> 4;                        // batch row
        const int off = (l & 15) << 7;                 // 128B line in row
        __builtin_prefetch(xn + (size_t)mm * ((size_t)T * I * sizeof(__bf16)) + off, 0, 1);
      }
    }

    v8f acc[4] = {};   // i, f, g, o gate tiles (16x16 f32)

    // gates += x_t @ w_ih^T
    for (int k = 0; k < I; k += 32) {
      v16bf a = ldfrag(pA1 + k);
#pragma unroll
      for (int g = 0; g < 4; ++g)
        acc[g] = wmma_bf16(a, ldfrag(pB1[g] + k), acc[g]);
    }
    // gates += h_{t-1} @ w_hh^T
    for (int k = 0; k < H; k += 32) {
      v16bf a = ldfrag(pa2 + k);
#pragma unroll
      for (int g = 0; g < 4; ++g)
        acc[g] = wmma_bf16(a, ldfrag(pB2[g] + k), acc[g]);
    }

    // Lane-local LSTM cell update.
#pragma unroll
    for (int e = 0; e < 8; ++e) {
      const int m = mbase + e;
      const float iv = fast_sigmoid(acc[0][e] + bi);
      const float fv = fast_sigmoid(acc[1][e] + bf);
      const float gv = fast_tanh(acc[2][e] + bg);
      const float ov = fast_sigmoid(acc[3][e] + bo);
      const float cv = fv * creg[e] + iv * gv;
      const float hv = ov * fast_tanh(cv);
      creg[e] = cv;
      hx[(size_t)m * H + n] = (__bf16)hv;
      out[(size_t)m * ((size_t)T * H) + (size_t)t * H + n] = hv;
      if (t == T - 1) {
        hn[(size_t)m * H + n] = hv;
        cn[(size_t)m * H + n] = cv;
      }
    }

    pA1 += I;                       // advance x to next time step

    if (t + 1 < T) grid_sync(bar, (unsigned)(t + 1));
  }
}

// ---------------------------------------------------------------------------
// Prep kernels
// ---------------------------------------------------------------------------
__global__ void cvt_bf16_kernel(const float* __restrict__ src,
                                __bf16* __restrict__ dst, long nelem) {
  long i = (long)blockIdx.x * blockDim.x + threadIdx.x;
  const long stride = (long)gridDim.x * blockDim.x;
  for (; i < nelem; i += stride) dst[i] = (__bf16)src[i];
}

__global__ void prep_misc_kernel(const float* __restrict__ bih,
                                 const float* __restrict__ bhh,
                                 float* __restrict__ bsum,
                                 __bf16* __restrict__ h0,
                                 __bf16* __restrict__ h1,
                                 unsigned* __restrict__ bar) {
  const int i = blockIdx.x * blockDim.x + threadIdx.x;
  if (i < G) bsum[i] = bih[i] + bhh[i];
  if (i < B * H) {
    h0[i] = (__bf16)0.0f;
    h1[i] = (__bf16)0.0f;
  }
  if (i == 0) { bar[0] = 0u; bar[1] = 0u; }
}

// ---------------------------------------------------------------------------
// Launch
// ---------------------------------------------------------------------------
extern "C" void kernel_launch(void* const* d_in, const int* in_sizes, int n_in,
                              void* d_out, int out_size, void* d_ws, size_t ws_size,
                              hipStream_t stream) {
  (void)in_sizes; (void)n_in; (void)out_size; (void)ws_size;

  const float* x   = (const float*)d_in[0];  // [B][T][I]
  const float* wih = (const float*)d_in[1];  // [G][I]
  const float* bih = (const float*)d_in[2];  // [G]
  const float* whh = (const float*)d_in[3];  // [G][H]
  const float* bhh = (const float*)d_in[4];  // [G]

  // workspace layout (~85 MB, 16B aligned throughout)
  char* ws = (char*)d_ws;
  size_t o = 0;
  __bf16* wihB = (__bf16*)(ws + o); o += (size_t)G * I * sizeof(__bf16);
  __bf16* whhB = (__bf16*)(ws + o); o += (size_t)G * H * sizeof(__bf16);
  __bf16* xB   = (__bf16*)(ws + o); o += (size_t)B * T * I * sizeof(__bf16);
  __bf16* h0   = (__bf16*)(ws + o); o += (size_t)B * H * sizeof(__bf16);
  __bf16* h1   = (__bf16*)(ws + o); o += (size_t)B * H * sizeof(__bf16);
  float*  bsum = (float*)(ws + o);  o += (size_t)G * sizeof(float);
  unsigned* bar = (unsigned*)(ws + o); o += 16;

  float* out = (float*)d_out;                 // [B][T][H]
  float* hn  = out + (size_t)B * T * H;       // [1][B][H]
  float* cn  = hn + (size_t)B * H;            // [1][B][H]

  // Parallel prep: bf16 conversions, bias sum, state + barrier init.
  cvt_bf16_kernel<<<4096, 256, 0, stream>>>(wih, wihB, (long)G * I);
  cvt_bf16_kernel<<<4096, 256, 0, stream>>>(whh, whhB, (long)G * H);
  cvt_bf16_kernel<<<8192, 256, 0, stream>>>(x, xB, (long)B * T * I);
  prep_misc_kernel<<<(B * H + 255) / 256, 256, 0, stream>>>(bih, bhh, bsum,
                                                            h0, h1, bar);

  // One persistent kernel runs the entire 512-step recurrence.
  lstm_persistent_kernel<<<NBLK, 256, 0, stream>>>(xB, wihB, whhB, bsum,
                                                   h0, h1, out, hn, cn, bar);
}
